// EpisodicMemory_52561809768514
// MI455X (gfx1250) — matrix operand order repaired and checked by reading, not compiled, not run on hardware
//
#include <hip/hip_runtime.h>
#include <math.h>

// ---------------------------------------------------------------------------
// Problem constants (from reference)
// ---------------------------------------------------------------------------
#define Bq   8
#define Sq   4096
#define Dq   1024
#define Mq   64
#define Hq   4
#define DHq  256          // Dq / Hq
#define BSq  (Bq * Sq)    // 32768

typedef __attribute__((ext_vector_type(16))) __bf16 v16bf;
typedef __attribute__((ext_vector_type(8)))  __bf16 v8bf;
typedef __attribute__((ext_vector_type(8)))  float  v8f;

// float -> bf16, round-to-nearest-even
__device__ __forceinline__ unsigned short f2bf(float f) {
    unsigned int u = __builtin_bit_cast(unsigned int, f);
    u += 0x7FFFu + ((u >> 16) & 1u);
    return (unsigned short)(u >> 16);
}

// Load one K-step worth of fragments (A: 16x32 tile, B: 4x 32x16 tiles).
__device__ __forceinline__ void loadFrags(const unsigned short* __restrict__ arow,
                                          const unsigned short* __restrict__ b,
                                          int j0, int ln, int hi, int ldb, int k,
                                          v16bf& af, v16bf bf[4]) {
    v8bf alo = *(const v8bf*)(arow + k + hi * 16);
    v8bf ahi = *(const v8bf*)(arow + k + hi * 16 + 8);
#pragma unroll
    for (int i = 0; i < 8; ++i) { af[i] = alo[i]; af[i + 8] = ahi[i]; }
#pragma unroll
    for (int nt = 0; nt < 4; ++nt)
        bf[nt] = *(const v16bf*)(b + (size_t)(j0 + nt * 16 + ln) * ldb + k + hi * 16);
}

__device__ __forceinline__ void mma4(const v16bf& af, const v16bf bf[4], v8f acc[4]) {
#pragma unroll
    for (int nt = 0; nt < 4; ++nt)
        acc[nt] = __builtin_amdgcn_wmma_f32_16x16x32_bf16(
            false, af, false, bf[nt], (short)0, acc[nt], false, false);
}

// ---------------------------------------------------------------------------
// Generic bf16 WMMA GEMM:  C[i,j] = alpha * (sum_k A[i,k]*B[j,k] + bias[j]) (+Cprev)
// Block: 256 threads = 8 waves. Wave tile: 16 rows x 64 cols (4 wmma accums).
// Block tile: 128 x 64.  Batched via blockIdx.z with (outer,inner) strides.
// K must be a multiple of 64 (two-stage software-pipelined inner loop).
// ---------------------------------------------------------------------------
template <bool OUT_BF16, bool ACC>
__global__ __launch_bounds__(256)
void gemmWmma(const unsigned short* __restrict__ A,
              const unsigned short* __restrict__ Bm,
              const float* __restrict__ bias, float alpha,
              void* __restrict__ C,
              int Mrows, int K, int lda, int ldb, int ldc,
              long aOut, long aIn, long bOut, long bIn, long cOut, long cIn,
              int zH)
{
    const int z  = blockIdx.z;
    const int zo = z / zH;
    const int zi = z - zo * zH;
    const unsigned short* a = A  + (size_t)zo * aOut + (size_t)zi * aIn;
    const unsigned short* b = Bm + (size_t)zo * bOut + (size_t)zi * bIn;
    const size_t cb = (size_t)zo * cOut + (size_t)zi * cIn;

    const int lane = threadIdx.x & 31;
    const int wave = threadIdx.x >> 5;
    const int i0 = blockIdx.y * 128 + wave * 16;
    if (i0 >= Mrows) return;                       // whole-wave exit, EXEC stays all-1s
    const int j0 = blockIdx.x * 64;
    const int hi = lane >> 4;
    const int ln = lane & 15;

    v8f acc[4];
#pragma unroll
    for (int nt = 0; nt < 4; ++nt)
#pragma unroll
        for (int r = 0; r < 8; ++r) acc[nt][r] = 0.0f;

    const unsigned short* arow = a + (size_t)(i0 + ln) * lda;

    // ---- two-stage software pipeline over K (K % 64 == 0) ----
    v16bf afA, afB;
    v16bf bfA[4], bfB[4];
    loadFrags(arow, b, j0, ln, hi, ldb, 0, afA, bfA);
    for (int k = 0; k < K; k += 64) {
        loadFrags(arow, b, j0, ln, hi, ldb, k + 32, afB, bfB);   // in flight over mma(A)
        if (k + 128 < K) __builtin_prefetch(arow + k + 128, 0, 1); // global_prefetch_b8
        mma4(afA, bfA, acc);
        if (k + 64 < K) loadFrags(arow, b, j0, ln, hi, ldb, k + 64, afA, bfA); // over mma(B)
        mma4(afB, bfB, acc);
    }

#pragma unroll
    for (int nt = 0; nt < 4; ++nt) {
        const int j  = j0 + nt * 16 + ln;
        const float bv = bias ? bias[j] : 0.0f;
#pragma unroll
        for (int r = 0; r < 8; ++r) {
            const size_t idx = cb + (size_t)(i0 + r + hi * 8) * ldc + j;
            float val = alpha * (acc[nt][r] + bv);
            if (OUT_BF16) {
                ((unsigned short*)C)[idx] = f2bf(val);
            } else {
                float prev = ACC ? ((float*)C)[idx] : 0.0f;
                ((float*)C)[idx] = val + prev;
            }
        }
    }
}

// ---------------------------------------------------------------------------
// Elementwise / helper kernels
// ---------------------------------------------------------------------------
__global__ void convertBf16(const float* __restrict__ in, unsigned short* __restrict__ out, long n) {
    long i = (long)blockIdx.x * blockDim.x + threadIdx.x;
    if (i < n) out[i] = f2bf(in[i]);
}

// f32 (R,C) -> bf16 (C,R), batched over blockIdx.z
__global__ void transposeF32Bf16(const float* __restrict__ in, unsigned short* __restrict__ out,
                                 int R, int C, long inB, long outB) {
    __shared__ float tile[32][33];
    const float* src = in  + (size_t)blockIdx.z * inB;
    unsigned short* dst = out + (size_t)blockIdx.z * outB;
    const int c0 = blockIdx.x * 32, r0 = blockIdx.y * 32;
    const int tx = threadIdx.x, ty = threadIdx.y;   // (32,8)
#pragma unroll
    for (int i = 0; i < 4; ++i) {
        int r = r0 + ty + i * 8;
        if (r < R && c0 + tx < C) tile[ty + i * 8][tx] = src[(size_t)r * C + c0 + tx];
    }
    __syncthreads();
#pragma unroll
    for (int i = 0; i < 4; ++i) {
        int c = c0 + ty + i * 8, r = r0 + tx;
        if (c < C && r < R) dst[(size_t)c * R + r] = f2bf(tile[tx][ty + i * 8]);
    }
}

// gate[row] = sigmoid(dot(x[row,:], wg) + wgb) ; one wave per row
__global__ void gateKernel(const float* __restrict__ x, const float* __restrict__ wg,
                           const float* __restrict__ wgb, float* __restrict__ gate, int rows) {
    const int lane = threadIdx.x & 31, wave = threadIdx.x >> 5;
    const int row = blockIdx.x * 8 + wave;
    if (row >= rows) return;
    const float* xr = x + (size_t)row * Dq;
    float s = 0.0f;
    for (int i = lane; i < Dq; i += 32) s += xr[i] * wg[i];
    for (int off = 16; off > 0; off >>= 1) s += __shfl_xor(s, off, 32);
    if (lane == 0) gate[row] = 1.0f / (1.0f + expf(-(s + wgb[0])));
}

// biasf[j] = comb_b[j] + dot(out_b, comb_w[j, D:2D]) ; one wave per row
__global__ void biasFuse(const float* __restrict__ combw, const float* __restrict__ outb,
                         const float* __restrict__ combb, float* __restrict__ biasf) {
    const int lane = threadIdx.x & 31, wave = threadIdx.x >> 5;
    const int j = blockIdx.x * 8 + wave;
    if (j >= Dq) return;
    const float* row = combw + (size_t)j * 2 * Dq + Dq;
    float s = 0.0f;
    for (int e = lane; e < Dq; e += 32) s += row[e] * outb[e];
    for (int off = 16; off > 0; off >>= 1) s += __shfl_xor(s, off, 32);
    if (lane == 0) biasf[j] = combb[j] + s;
}

// in-place: sim row (64) -> softmax * gate[row]
__global__ void softmaxGate(float* __restrict__ sg, const float* __restrict__ gate, int rows) {
    const int row = blockIdx.x * blockDim.x + threadIdx.x;
    if (row >= rows) return;
    float* p = sg + (size_t)row * Mq;
    float v[Mq], mx = -1e30f;
#pragma unroll 8
    for (int i = 0; i < Mq; ++i) { v[i] = p[i]; mx = fmaxf(mx, v[i]); }
    float sum = 0.0f;
#pragma unroll 8
    for (int i = 0; i < Mq; ++i) { v[i] = expf(v[i] - mx); sum += v[i]; }
    const float g = gate[row] / sum;
#pragma unroll 8
    for (int i = 0; i < Mq; ++i) p[i] = v[i] * g;
}

// scores row (64) -> softmax -> bf16
__global__ void softmaxP(const float* __restrict__ sc, unsigned short* __restrict__ pb, long rows) {
    const long row = (long)blockIdx.x * blockDim.x + threadIdx.x;
    if (row >= rows) return;
    const float* p = sc + (size_t)row * Mq;
    float v[Mq], mx = -1e30f;
#pragma unroll 8
    for (int i = 0; i < Mq; ++i) { v[i] = p[i]; mx = fmaxf(mx, v[i]); }
    float sum = 0.0f;
#pragma unroll 8
    for (int i = 0; i < Mq; ++i) { v[i] = expf(v[i] - mx); sum += v[i]; }
    const float inv = 1.0f / sum;
    unsigned short* o = pb + (size_t)row * Mq;
#pragma unroll 8
    for (int i = 0; i < Mq; ++i) o[i] = f2bf(v[i] * inv);
}

// slotg[b,m] = min(sum_s gated[b,s,m], 1) ; one block per batch
__global__ void slotSum(const float* __restrict__ gated, float* __restrict__ slotg) {
    const int b = blockIdx.x, t = threadIdx.x;
    const int m = t & 63, sr = t >> 6;          // 4 s-lanes per m
    const float* g = gated + (size_t)b * Sq * Mq;
    float acc = 0.0f;
    for (int s = sr; s < Sq; s += 4) acc += g[(size_t)s * Mq + m];
    __shared__ float red[256];
    red[t] = acc;
    __syncthreads();
    if (t < 64) {
        float tot = red[t] + red[t + 64] + red[t + 128] + red[t + 192];
        slotg[b * Mq + m] = fminf(tot, 1.0f);
    }
}

// mv_bf16[b,m,d] = bf16(slotg[b,m] * wv[b,m,d])
__global__ void scaleConv(const float* __restrict__ wv, const float* __restrict__ slotg,
                          unsigned short* __restrict__ mv, long n) {
    long i = (long)blockIdx.x * blockDim.x + threadIdx.x;
    if (i < n) {
        long bm = i / Dq;
        mv[i] = f2bf(slotg[bm] * wv[i]);
    }
}

// ---------------------------------------------------------------------------
// Host-side launcher
// ---------------------------------------------------------------------------
extern "C" void kernel_launch(void* const* d_in, const int* in_sizes, int n_in,
                              void* d_out, int out_size, void* d_ws, size_t ws_size,
                              hipStream_t stream) {
    (void)in_sizes; (void)n_in; (void)out_size; (void)ws_size;

    const float* x     = (const float*)d_in[0];
    const float* mk    = (const float*)d_in[1];
    const float* wgw   = (const float*)d_in[2];
    const float* wgb   = (const float*)d_in[3];
    const float* inpw  = (const float*)d_in[4];
    const float* inpb  = (const float*)d_in[5];
    const float* outw  = (const float*)d_in[6];
    const float* outb  = (const float*)d_in[7];
    const float* combw = (const float*)d_in[8];
    const float* combb = (const float*)d_in[9];
    float* out = (float*)d_out;

    // ---- workspace carve-up (256B aligned) ----
    char* base = (char*)d_ws;
    size_t off = 0;
    auto carve = [&](size_t bytes) -> char* {
        char* p = base + off;
        off = (off + bytes + 255) & ~(size_t)255;
        return p;
    };
    unsigned short* xbf    = (unsigned short*)carve((size_t)BSq * Dq * 2);
    unsigned short* wbf    = (unsigned short*)carve((size_t)3 * Dq * Dq * 2);
    unsigned short* mkbf   = (unsigned short*)carve((size_t)Mq * Dq * 2);
    unsigned short* combbf = (unsigned short*)carve((size_t)Dq * 2 * Dq * 2);
    unsigned short* outwT  = (unsigned short*)carve((size_t)Dq * Dq * 2);
    unsigned short* Wfbf   = (unsigned short*)carve((size_t)Dq * Dq * 2);
    float*          biasf  = (float*)carve((size_t)Dq * 4);
    float*          gate   = (float*)carve((size_t)BSq * 4);
    float*          simg   = (float*)carve((size_t)BSq * Mq * 4);      // sim, then gated (in place)
    unsigned short* gatedT = (unsigned short*)carve((size_t)Bq * Mq * Sq * 2);
    unsigned short* xT     = (unsigned short*)carve((size_t)Bq * Dq * Sq * 2); // later reused as mem_out
    float*          slotg  = (float*)carve((size_t)Bq * Mq * 4);
    float*          wv     = (float*)carve((size_t)Bq * Mq * Dq * 4);
    unsigned short* mv     = (unsigned short*)carve((size_t)Bq * Mq * Dq * 2);
    unsigned short* qbf    = (unsigned short*)carve((size_t)BSq * Dq * 2);
    unsigned short* kbf    = (unsigned short*)carve((size_t)Mq * Dq * 2);
    float*          vf     = (float*)carve((size_t)Bq * Mq * Dq * 4);
    unsigned short* vT     = (unsigned short*)carve((size_t)Bq * Dq * Mq * 2);
    float*          scores = (float*)carve((size_t)Bq * Hq * Sq * Mq * 4);
    unsigned short* pbf    = (unsigned short*)carve((size_t)Bq * Hq * Sq * Mq * 2);
    unsigned short* memout = xT;   // xT is dead after write_vals; reuse for mem_out (same size)

    const dim3 blk256(256);

    // 1) bf16 conversions of x and weights
    {
        long n = (long)BSq * Dq;
        convertBf16<<<dim3((n + 255) / 256), blk256, 0, stream>>>(x, xbf, n);
        n = (long)3 * Dq * Dq;
        convertBf16<<<dim3((n + 255) / 256), blk256, 0, stream>>>(inpw, wbf, n);
        n = (long)Mq * Dq;
        convertBf16<<<dim3((n + 255) / 256), blk256, 0, stream>>>(mk, mkbf, n);
        n = (long)Dq * 2 * Dq;
        convertBf16<<<dim3((n + 255) / 256), blk256, 0, stream>>>(combw, combbf, n);
    }
    // out_w^T (d,e) bf16 for the weight-fusion GEMM
    transposeF32Bf16<<<dim3(Dq / 32, Dq / 32, 1), dim3(32, 8), 0, stream>>>(
        outw, outwT, Dq, Dq, 0, 0);
    // fused bias: comb_b + comb_w[:,D:] @ out_b
    biasFuse<<<dim3(Dq / 8), blk256, 0, stream>>>(combw, outb, combb, biasf);
    // write gate
    gateKernel<<<dim3(BSq / 8), blk256, 0, stream>>>(x, wgw, wgb, gate, BSq);

    // 2) sim = x @ mk^T   (BS x 64, K=D)
    gemmWmma<false, false><<<dim3(1, BSq / 128, 1), blk256, 0, stream>>>(
        xbf, mkbf, nullptr, 1.0f, simg, BSq, Dq, Dq, Dq, Mq, 0, 0, 0, 0, 0, 0, 1);
    // 3) gated = softmax(sim) * gate   (in place)
    softmaxGate<<<dim3(BSq / 256), blk256, 0, stream>>>(simg, gate, BSq);
    // 4) transposes for write_vals
    transposeF32Bf16<<<dim3(Mq / 32, Sq / 32, Bq), dim3(32, 8), 0, stream>>>(
        simg, gatedT, Sq, Mq, (long)Sq * Mq, (long)Mq * Sq);
    transposeF32Bf16<<<dim3(Dq / 32, Sq / 32, Bq), dim3(32, 8), 0, stream>>>(
        x, xT, Sq, Dq, (long)Sq * Dq, (long)Dq * Sq);
    // 5) slot gate
    slotSum<<<dim3(Bq), blk256, 0, stream>>>(simg, slotg);
    // 6) write_vals[b] = gated^T @ x   (64 x D, K=S), batch b
    gemmWmma<false, false><<<dim3(Dq / 64, 1, Bq), blk256, 0, stream>>>(
        gatedT, xT, nullptr, 1.0f, wv, Mq, Sq, Sq, Sq, Dq,
        (long)Mq * Sq, 0, (long)Dq * Sq, 0, (long)Mq * Dq, 0, 1);
    // 7) memory_values (bf16) = slot_gate * write_vals
    {
        long n = (long)Bq * Mq * Dq;
        scaleConv<<<dim3((n + 255) / 256), blk256, 0, stream>>>(wv, slotg, mv, n);
    }
    // 8) q = (x @ wq^T + bq) / 16  -> bf16     (BS x D, K=D)
    gemmWmma<true, false><<<dim3(Dq / 64, BSq / 128, 1), blk256, 0, stream>>>(
        xbf, wbf, inpb, 1.0f / 16.0f, qbf, BSq, Dq, Dq, Dq, Dq, 0, 0, 0, 0, 0, 0, 1);
    // 9) k = mk @ wk^T + bk -> bf16            (64 x D, K=D)
    gemmWmma<true, false><<<dim3(Dq / 64, 1, 1), blk256, 0, stream>>>(
        mkbf, wbf + (size_t)Dq * Dq, inpb + Dq, 1.0f, kbf, Mq, Dq, Dq, Dq, Dq,
        0, 0, 0, 0, 0, 0, 1);
    // 10) v[b] = mv[b] @ wv^T + bv -> f32      (64 x D, K=D), batch b
    gemmWmma<false, false><<<dim3(Dq / 64, 1, Bq), blk256, 0, stream>>>(
        mv, wbf + (size_t)2 * Dq * Dq, inpb + 2 * Dq, 1.0f, vf, Mq, Dq, Dq, Dq, Dq,
        (long)Mq * Dq, 0, 0, 0, (long)Mq * Dq, 0, 1);
    // 11) v^T (b: D x M) bf16
    transposeF32Bf16<<<dim3(Dq / 32, Mq / 32, Bq), dim3(32, 8), 0, stream>>>(
        vf, vT, Mq, Dq, (long)Mq * Dq, (long)Dq * Mq);
    // 12) fused weight Wf = comb_w[:,D:] @ out_w  -> bf16  (D x D, K=D)
    gemmWmma<true, false><<<dim3(Dq / 64, Dq / 128, 1), blk256, 0, stream>>>(
        combbf + Dq, outwT, nullptr, 1.0f, Wfbf, Dq, Dq, 2 * Dq, Dq, Dq,
        0, 0, 0, 0, 0, 0, 1);
    // 13) scores[b,h] = qh @ kh^T   (S x 64, K=DH), z = b*H + h
    gemmWmma<false, false><<<dim3(1, Sq / 128, Bq * Hq), blk256, 0, stream>>>(
        qbf, kbf, nullptr, 1.0f, scores, Sq, DHq, Dq, Dq, Mq,
        (long)Sq * Dq, (long)DHq, 0, (long)DHq, (long)Hq * Sq * Mq, (long)Sq * Mq, Hq);
    // 14) p = softmax(scores) -> bf16
    {
        long rows = (long)Bq * Hq * Sq;
        softmaxP<<<dim3((rows + 255) / 256), blk256, 0, stream>>>(scores, pbf, rows);
    }
    // 15) mem_out[b,:,h*DH:...] = p[b,h] @ v[b,:,h]^T -> bf16  (S x DH, K=64)
    gemmWmma<true, false><<<dim3(DHq / 64, Sq / 128, Bq * Hq), blk256, 0, stream>>>(
        pbf, vT, nullptr, 1.0f, memout, Sq, Mq, Mq, Mq, Dq,
        (long)Hq * Sq * Mq, (long)Sq * Mq, (long)Dq * Mq, (long)DHq * Mq,
        (long)Sq * Dq, (long)DHq, Hq);
    // 16) out  = x @ comb_w[:,:D]^T + biasf       (BS x D, K=D)
    gemmWmma<false, false><<<dim3(Dq / 64, BSq / 128, 1), blk256, 0, stream>>>(
        xbf, combbf, biasf, 1.0f, out, BSq, Dq, Dq, 2 * Dq, Dq, 0, 0, 0, 0, 0, 0, 1);
    // 17) out += mem_out @ Wf^T                   (BS x D, K=D, accumulate)
    gemmWmma<false, true><<<dim3(Dq / 64, BSq / 128, 1), blk256, 0, stream>>>(
        memout, Wfbf, nullptr, 1.0f, out, BSq, Dq, Dq, Dq, Dq, 0, 0, 0, 0, 0, 0, 1);
}
